// PerslayModel_7481833030230
// MI455X (gfx1250) — matrix operand order repaired and checked by reading, not compiled, not run on hardware
//
#include <hip/hip_runtime.h>
#include <hip/hip_bf16.h>

typedef __attribute__((ext_vector_type(16))) _Float16 v16h;
typedef __attribute__((ext_vector_type(8)))  float    v8f;

#define B_DIM 64
#define N_PTS 2048
#define G_DIM 32
#define K_DIM 512
#define F_DIM 128
#define OUT_STRIDE (K_DIM + F_DIM)   // 640

// ---------------------------------------------------------------------------
// Pass 1: s[b,n] = mask * sum_g exp(-((x-Mgx)^2*Vgx^2 + (y-Mgy)^2*Vgy^2))
// ---------------------------------------------------------------------------
__global__ __launch_bounds__(256) void perslay_weight_kernel(
    const float* __restrict__ diag,   // (B, N, 3)
    const float* __restrict__ Wg,     // (4, G)
    float* __restrict__ s_ws)         // (B*N)
{
    __shared__ float sh_w[4 * G_DIM];
    const int tid = threadIdx.x;
    if (tid < 4 * G_DIM) sh_w[tid] = Wg[tid];
    __syncthreads();

    const int idx = blockIdx.x * 256 + tid;           // over B*N
    if (idx >= B_DIM * N_PTS) return;

    const float x = diag[idx * 3 + 0];
    const float y = diag[idx * 3 + 1];
    const float m = diag[idx * 3 + 2];

    float w = 0.0f;
#pragma unroll
    for (int g = 0; g < G_DIM; ++g) {
        const float dx = x - sh_w[0 * G_DIM + g];
        const float dy = y - sh_w[1 * G_DIM + g];
        const float vx = sh_w[2 * G_DIM + g];
        const float vy = sh_w[3 * G_DIM + g];
        w += __expf(-(dx * dx * vx * vx + dy * dy * vy * vy));
    }
    s_ws[idx] = w * m;
}

// ---------------------------------------------------------------------------
// Pass 2: vec[b, k0:k0+16] via WMMA contraction over N.
//   D = A x B + C, 16x16x32 f16->f32.
//   Contraction dim (Kc=32) = chunk of 32 points n.
//   B (32x16): B[j,kcol] = feat(n = chunk+j, k = k0+kcol). Lane L owns column
//     kcol = L%16; element e holds row j = e + 16*(L>=16).
//   A (16x32): only row m=0 is nonzero, carrying s[n]. Row 0 lives in lanes 0
//     and 16; matching B's k order: lane0 e -> K=e;
//     lane16 e -> K=e+16.
//   C row 0 (VGPR0, lanes 0..15) = partial vec[k0+lane].
// ---------------------------------------------------------------------------
__global__ __launch_bounds__(256) void perslay_vec_kernel(
    const float* __restrict__ diag,   // (B, N, 3)
    const float* __restrict__ LM,     // (2, K)
    const float* __restrict__ LV,     // (2, K)
    const float* __restrict__ LA,     // (K)
    const float* __restrict__ s_ws,   // (B*N)
    float* __restrict__ out)          // (B, 640)
{
    __shared__ float    sh_x[N_PTS];
    __shared__ float    sh_y[N_PTS];
    __shared__ _Float16 sh_s[N_PTS];
    __shared__ float    sh_red[8][16];

    const int b    = blockIdx.y;
    const int k0   = blockIdx.x * 16;
    const int tid  = threadIdx.x;
    const int lane = tid & 31;
    const int wave = tid >> 5;
    const int half = lane >> 4;       // 0: lanes 0-15, 1: lanes 16-31

    // Stage this batch's points into LDS (x, y in f32; s pre-converted to f16)
    for (int n = tid; n < N_PTS; n += 256) {
        sh_x[n] = diag[(b * N_PTS + n) * 3 + 0];
        sh_y[n] = diag[(b * N_PTS + n) * 3 + 1];
        sh_s[n] = (_Float16)s_ws[b * N_PTS + n];
    }
    __syncthreads();

    // Per-lane k parameters (loop-invariant: each lane owns one k column)
    const int   kl  = k0 + (lane & 15);
    const float lmx = LM[kl];
    const float lmy = LM[K_DIM + kl];
    const float lvx = fabsf(LV[kl]);
    const float lvy = fabsf(LV[K_DIM + kl]);
    const float la  = LA[kl];

    v8f c = {};

    for (int chunk = wave * 32; chunk < N_PTS; chunk += 256) {
        // ---- B fragment: 16 feature evaluations per lane (f32 math, f16 pack)
        v16h bf;
#pragma unroll
        for (int e = 0; e < 16; ++e) {
            const int   n    = chunk + e + 16 * half;
            const float x    = sh_x[n];
            const float y    = sh_y[n];
            const float base = 1.0f + fabsf(x - lmx) * lvx + fabsf(y - lmy) * lvy;
            const float f    = __expf(-la * __logf(base));   // base^(-la)
            bf[e] = (_Float16)f;
        }

        // ---- A fragment: s[n] in row m=0 only (lanes 0 and 16)
        v16h af;
#pragma unroll
        for (int e = 0; e < 16; ++e) af[e] = (_Float16)0.0f;
        if ((lane & 15) == 0) {
#pragma unroll
            for (int e = 0; e < 16; ++e) {
                const int Kc = e + 16 * half;
                af[e] = sh_s[chunk + Kc];
            }
        }

        // D(16x16,f32) = A(16x32,f16) x B(32x16,f16) + C
        c = __builtin_amdgcn_wmma_f32_16x16x32_f16(
                /*neg_a=*/false, af, /*neg_b=*/false, bf,
                /*c_mod=*/(short)0, c, /*reuse_a=*/false, /*reuse_b=*/false);
    }

    // Row m=0 of C: VGPR0 on lanes 0..15 holds N=lane
    if (lane < 16) sh_red[wave][lane] = c[0];
    __syncthreads();

    if (tid < 16) {
        float acc = 0.0f;
#pragma unroll
        for (int w = 0; w < 8; ++w) acc += sh_red[w][tid];
        out[b * OUT_STRIDE + k0 + tid] = acc;
    }
}

// ---------------------------------------------------------------------------
// Pass 3: out[b, 512 + f] = feats[b, f]
// ---------------------------------------------------------------------------
__global__ __launch_bounds__(256) void perslay_copy_feats(
    const float* __restrict__ feats,  // (B, F)
    float* __restrict__ out)          // (B, 640)
{
    const int idx = blockIdx.x * 256 + threadIdx.x;   // over B*F
    if (idx >= B_DIM * F_DIM) return;
    const int b = idx / F_DIM;
    const int f = idx % F_DIM;
    out[b * OUT_STRIDE + K_DIM + f] = feats[idx];
}

// ---------------------------------------------------------------------------
extern "C" void kernel_launch(void* const* d_in, const int* in_sizes, int n_in,
                              void* d_out, int out_size, void* d_ws, size_t ws_size,
                              hipStream_t stream) {
    const float* diag  = (const float*)d_in[0];   // (B, N, 3)
    const float* feats = (const float*)d_in[1];   // (B, F)
    const float* Wg    = (const float*)d_in[2];   // (4, G)
    const float* LM    = (const float*)d_in[3];   // (2, K)
    const float* LV    = (const float*)d_in[4];   // (2, K)
    const float* LA    = (const float*)d_in[5];   // (K)
    float*       out   = (float*)d_out;           // (B, 640)
    float*       s_ws  = (float*)d_ws;            // (B*N) floats = 512 KB

    (void)in_sizes; (void)n_in; (void)out_size; (void)ws_size;

    // Pass 1: per-point Gaussian weights
    perslay_weight_kernel<<<(B_DIM * N_PTS) / 256, 256, 0, stream>>>(diag, Wg, s_ws);

    // Pass 2: WMMA contraction -> vec
    dim3 grid2(K_DIM / 16, B_DIM);
    perslay_vec_kernel<<<grid2, 256, 0, stream>>>(diag, LM, LV, LA, s_ws, out);

    // Pass 3: passthrough features
    perslay_copy_feats<<<(B_DIM * F_DIM) / 256, 256, 0, stream>>>(feats, out);
}
